// PredictorLSTM_52321291599931
// MI455X (gfx1250) — matrix operand order repaired and checked by reading, not compiled, not run on hardware
//
#include <hip/hip_runtime.h>
#include <hip/hip_bf16.h>
#include <cstdint>
#include <cstddef>

// ---------------------------------------------------------------------------
// PredictorLSTM for MI455X (gfx1250).
//  - [W_ih|W_hh] concatenated -> 8192 x 2176 f16 matrix, LDS-resident:
//    128 persistent workgroups; block B owns the i/f/g/o gate rows of hidden
//    units [16B,16B+16): 4 tiles x 16 rows x 2176 x 2B = 272KB of the WGP's
//    320KB LDS. Weights never leave LDS after the one-time preload.
//  - matvec via v_wmma_f32_16x16x32_f16 with all 16 B-columns = [x_t ; h]
//    (M=1 batched redundantly into the otherwise idle WMMA pipe).
//  - 4 independent accumulators break the D->C dependency chain so the 68
//    WMMAs per step pipeline instead of serializing (latency is the metric
//    for a sequential recurrence).
//  - gates + cell state c stay block-local (LDS); h is double-buffered in
//    global memory so ONE device-scope barrier per step is sufficient.
// ---------------------------------------------------------------------------

typedef _Float16 f16;
typedef _Float16 v8h  __attribute__((ext_vector_type(8)));
typedef _Float16 v16h __attribute__((ext_vector_type(16)));
typedef float    v8f  __attribute__((ext_vector_type(8)));

#define T_STEPS 8192
#define F_IN    128
#define H_DIM   2048
#define G4H     (4 * H_DIM)        // 8192 gate rows [i|f|g|o]
#define KC      (F_IN + H_DIM)     // 2176 concatenated K
#define NCHUNK  (KC / 32)          // 68 WMMA K-chunks of 32
#define NBLK    128                // persistent blocks (1 per WGP)
#define NTHR    128                // 4 waves/block; wave w = gate type w

// ---------------- grid-wide barrier (device scope, sense via generation) ----
__device__ __forceinline__ void grid_barrier(unsigned* bar, unsigned nblocks) {
    __threadfence();           // make this block's global writes visible
    __syncthreads();
    if (threadIdx.x == 0) {
        unsigned* cnt = bar;
        unsigned* gen = bar + 1;
        unsigned g = __hip_atomic_load(gen, __ATOMIC_ACQUIRE, __HIP_MEMORY_SCOPE_AGENT);
        unsigned prev = __hip_atomic_fetch_add(cnt, 1u, __ATOMIC_ACQ_REL, __HIP_MEMORY_SCOPE_AGENT);
        if (prev == nblocks - 1u) {
            __hip_atomic_store(cnt, 0u, __ATOMIC_RELAXED, __HIP_MEMORY_SCOPE_AGENT);
            __hip_atomic_fetch_add(gen, 1u, __ATOMIC_ACQ_REL, __HIP_MEMORY_SCOPE_AGENT);
        } else {
            while (__hip_atomic_load(gen, __ATOMIC_ACQUIRE, __HIP_MEMORY_SCOPE_AGENT) == g) {
                __builtin_amdgcn_s_sleep(1);   // ~64clk wake granularity
            }
        }
    }
    __syncthreads();
}

// Straight-line fast activations (on the sequential critical path 8192x).
__device__ __forceinline__ float fsigmoid(float v) {
    return __fdividef(1.0f, 1.0f + __expf(-v));
}
__device__ __forceinline__ float ftanh(float v) {
    return __fdividef(2.0f, 1.0f + __expf(-2.0f * v)) - 1.0f;
}

// ---------------- K0: per-launch state init (ws is poisoned between calls) --
__global__ void k_init(const float* __restrict__ b_ih, const float* __restrict__ b_hh,
                       float* b, float* hf, f16* hh, unsigned* bar) {
    const int i = blockIdx.x * blockDim.x + threadIdx.x;
    if (i < G4H) b[i] = b_ih[i] + b_hh[i];
    if (i < 2 * H_DIM) hh[i] = (f16)0.0f;     // both h parity buffers
    if (i < H_DIM) hf[i] = 0.0f;
    if (i == 0) { bar[0] = 0u; bar[1] = 0u; }
}

// ---------------- K1: fp32 -> f16 staging of x and [W_ih | W_hh] -----------
__global__ void k_convert(const float* __restrict__ x, const float* __restrict__ W_ih,
                          const float* __restrict__ W_hh,
                          f16* __restrict__ x_h, f16* __restrict__ Wcat) {
    const size_t stride = (size_t)gridDim.x * blockDim.x;
    const size_t XT  = (size_t)T_STEPS * F_IN;
    const size_t TOT = XT + (size_t)G4H * KC;
    for (size_t i = (size_t)blockIdx.x * blockDim.x + threadIdx.x; i < TOT; i += stride) {
        if (i < XT) {
            x_h[i] = (f16)x[i];
        } else {
            size_t e = i - XT;
            size_t r = e / KC;
            size_t k = e % KC;
            float v = (k < F_IN) ? W_ih[r * F_IN + k] : W_hh[r * H_DIM + (k - F_IN)];
            Wcat[e] = (f16)v;
        }
    }
}

// A-operand lane layout: 16 contiguous K values K[k0+hs*16..+15]
// (matches the B operand order).
__device__ __forceinline__ v16h load_a16(const f16* arow, int k0, int hs) {
    v8h lo = *(const v8h*)(arow + k0 + hs * 16);
    v8h hi = *(const v8h*)(arow + k0 + hs * 16 + 8);
    return __builtin_shufflevector(lo, hi,
                                   0,1,2,3,4,5,6,7,8,9,10,11,12,13,14,15);
}

// ---------------- K2: persistent LSTM recurrence ---------------------------
__global__ void __launch_bounds__(NTHR, 1)
k_lstm(const f16* __restrict__ x_h, const f16* __restrict__ Wcat,
       const float* __restrict__ b,
       float* hf, f16* hh, unsigned* bar) {
    // 272KB weight slice + 4.25KB input vector + tiny gate/cell state.
    __shared__ __attribute__((aligned(32))) f16 wlds[4 * 16 * KC];
    __shared__ __attribute__((aligned(32))) f16 vin[KC];   // [x_t ; h_{t-1}]
    __shared__ float gl[64];    // this block's 4x16 gate pre-activations
    __shared__ float c_l[16];   // this block's cell state (never leaves LDS)

    const int tid  = threadIdx.x;
    const int wvb  = tid >> 5;          // wave within block == gate type 0..3
    const int lane = tid & 31;
    const int lrow = lane & 15;         // M row / N col within 16x16 tile
    const int hs   = lane >> 4;         // K-half selector per ISA layout

    // One-time preload: 4 slabs of 16 rows (gate g rows g*H + 16*B ..+15).
    {
        const int slab_u4 = 16 * KC * (int)sizeof(f16) / 16;   // 4352 uint4
        for (int w = 0; w < 4; ++w) {
            const uint4* src = (const uint4*)(Wcat +
                ((size_t)w * H_DIM + (size_t)blockIdx.x * 16) * KC);
            uint4* dst = (uint4*)(wlds + (size_t)w * 16 * KC);
            for (int i = tid; i < slab_u4; i += NTHR) dst[i] = src[i];
        }
        if (tid < 16) c_l[tid] = 0.0f;
    }
    __syncthreads();

    // Per-lane bias folded into accumulator 0's init (only D column 0 is
    // extracted, so loading b[row] into every lane's acc is valid).
    v8f bias;
    {
        const float* bp = b + (size_t)wvb * H_DIM + (size_t)blockIdx.x * 16 + hs * 8;
        #pragma unroll
        for (int r = 0; r < 8; ++r) bias[r] = bp[r];
    }

    const f16* arow = &wlds[((size_t)wvb * 16 + lrow) * KC];
    const v8f vzero = {0.f, 0.f, 0.f, 0.f, 0.f, 0.f, 0.f, 0.f};

    for (int t = 0; t < T_STEPS; ++t) {
        // Stage [x_t ; h_{t-1}] into LDS as b128 copies (16 + 256 uint4).
        {
            const uint4* sx = (const uint4*)(x_h + (size_t)t * F_IN);
            const uint4* sh = (const uint4*)(hh + (size_t)(t & 1) * H_DIM);
            uint4* dv = (uint4*)vin;
            const int NX = F_IN * (int)sizeof(f16) / 16;            // 16
            const int NT = KC * (int)sizeof(f16) / 16;              // 272
            for (int i = tid; i < NT; i += NTHR)
                dv[i] = (i < NX) ? sx[i] : sh[i - NX];
        }
        __syncthreads();

        // gate_pre[16] = bias + Wtile (16 x 2176) * [x_t ; h]
        // 4 independent accumulators -> no D->C serialization.
        v8f acc0 = bias, acc1 = vzero, acc2 = vzero, acc3 = vzero;
        #pragma unroll 1
        for (int kc = 0; kc < NCHUNK; kc += 4) {
            const int k0 = kc * 32;
            v16h a0 = load_a16(arow, k0,      hs);
            v16h a1 = load_a16(arow, k0 + 32, hs);
            v16h a2 = load_a16(arow, k0 + 64, hs);
            v16h a3 = load_a16(arow, k0 + 96, hs);
            // B lane layout (32x16): 16 contiguous K values at k0 + hs*16.
            v16h b0 = *(const v16h*)(&vin[k0      + hs * 16]);
            v16h b1 = *(const v16h*)(&vin[k0 + 32 + hs * 16]);
            v16h b2 = *(const v16h*)(&vin[k0 + 64 + hs * 16]);
            v16h b3 = *(const v16h*)(&vin[k0 + 96 + hs * 16]);
            acc0 = __builtin_amdgcn_wmma_f32_16x16x32_f16(
                       false, a0, false, b0, (short)0, acc0, false, false);
            acc1 = __builtin_amdgcn_wmma_f32_16x16x32_f16(
                       false, a1, false, b1, (short)0, acc1, false, false);
            acc2 = __builtin_amdgcn_wmma_f32_16x16x32_f16(
                       false, a2, false, b2, (short)0, acc2, false, false);
            acc3 = __builtin_amdgcn_wmma_f32_16x16x32_f16(
                       false, a3, false, b3, (short)0, acc3, false, false);
        }
        v8f acc = (acc0 + acc1) + (acc2 + acc3);

        // D layout: column N==0 lives in lanes 0 (M 0..7) and 16 (M 8..15).
        if (lrow == 0) {
            #pragma unroll
            for (int r = 0; r < 8; ++r) gl[wvb * 16 + hs * 8 + r] = acc[r];
        }
        __syncthreads();

        // Block-local cell update for hidden units [16*B, 16*B+16).
        if (tid < 16) {
            float iv = fsigmoid(gl[tid]);
            float fv = fsigmoid(gl[16 + tid]);
            float gv = ftanh(gl[32 + tid]);
            float ov = fsigmoid(gl[48 + tid]);
            float cn = fv * c_l[tid] + iv * gv;
            c_l[tid] = cn;
            float hn = ov * ftanh(cn);
            const int jg = blockIdx.x * 16 + tid;
            hh[(size_t)((t + 1) & 1) * H_DIM + jg] = (f16)hn;  // write parity
            if (t == T_STEPS - 1) hf[jg] = hn;                 // fp32 for head
        }

        // Single device-scope barrier per step: publishes h(t) before any
        // block stages it at step t+1 (double buffering kills the WAR race).
        grid_barrier(bar, gridDim.x);
    }
}

// ---------------- K3: y = (h W_lin^T + b_lin) W_out^T + b_out --------------
__global__ void k_head(const float* __restrict__ hf,
                       const float* __restrict__ W_lin, const float* __restrict__ b_lin,
                       const float* __restrict__ W_out, const float* __restrict__ b_out,
                       float* out) {
    __shared__ float y1[H_DIM];
    __shared__ float red[256];
    const int tid = threadIdx.x;
    for (int e = tid; e < H_DIM; e += 256) {
        const float* wr = W_lin + (size_t)e * H_DIM;
        float s = b_lin[e];
        for (int k = 0; k < H_DIM; ++k) s += hf[k] * wr[k];
        y1[e] = s;
    }
    __syncthreads();
    for (int r = 0; r < 2; ++r) {
        float p = 0.0f;
        for (int k = tid; k < H_DIM; k += 256) p += y1[k] * W_out[r * H_DIM + k];
        red[tid] = p;
        __syncthreads();
        for (int s = 128; s > 0; s >>= 1) {
            if (tid < s) red[tid] += red[tid + s];
            __syncthreads();
        }
        if (tid == 0) out[r] = red[0] + b_out[r];
        __syncthreads();
    }
}

// ---------------------------------------------------------------------------
extern "C" void kernel_launch(void* const* d_in, const int* in_sizes, int n_in,
                              void* d_out, int out_size, void* d_ws, size_t ws_size,
                              hipStream_t stream) {
    (void)in_sizes; (void)n_in; (void)out_size; (void)ws_size;

    const float* x     = (const float*)d_in[0];
    const float* W_ih  = (const float*)d_in[1];
    const float* W_hh  = (const float*)d_in[2];
    const float* b_ih  = (const float*)d_in[3];
    const float* b_hh  = (const float*)d_in[4];
    const float* W_lin = (const float*)d_in[5];
    const float* b_lin = (const float*)d_in[6];
    const float* W_out = (const float*)d_in[7];
    const float* b_out = (const float*)d_in[8];
    float* out = (float*)d_out;

    // Workspace carve (~38 MB), 256B aligned slices.
    char* cur = (char*)d_ws;
    auto carve = [&](size_t bytes) -> char* {
        char* p = cur;
        cur += (bytes + 255) & ~(size_t)255;
        return p;
    };
    unsigned* bar  = (unsigned*)carve(256);
    float*    b    = (float*)carve(sizeof(float) * G4H);
    float*    hf   = (float*)carve(sizeof(float) * H_DIM);
    f16*      hh   = (f16*)carve(sizeof(f16) * 2 * H_DIM);   // double buffered
    f16*      x_h  = (f16*)carve(sizeof(f16) * (size_t)T_STEPS * F_IN);
    f16*      Wcat = (f16*)carve(sizeof(f16) * (size_t)G4H * KC);

    k_init<<<dim3(G4H / 256), dim3(256), 0, stream>>>(b_ih, b_hh, b, hf, hh, bar);
    k_convert<<<dim3(1024), dim3(256), 0, stream>>>(x, W_ih, W_hh, x_h, Wcat);
    k_lstm<<<dim3(NBLK), dim3(NTHR), 0, stream>>>(x_h, Wcat, b, hf, hh, bar);
    k_head<<<dim3(1), dim3(256), 0, stream>>>(hf, W_lin, b_lin, W_out, b_out, out);
}
